// LocalSensitiveAttention_8804682957184
// MI455X (gfx1250) — compile-verified, matrix operand order checked by reading, not run on hardware
//
#include <hip/hip_runtime.h>

// Problem constants (from reference)
#define BB      64
#define TT_     2048
#define DEC     1024
#define ENC     512
#define ATT     128
#define LOC     32
#define KW      31
#define PADW    15

// Tiling
#define S_SPLIT 8                    // T-chunks per batch -> grid 64*8 = 512 WGs
#define T_CHUNK (TT_ / S_SPLIT)      // 256
#define T_TILE  32
#define N_TILES (T_CHUNK / T_TILE)   // 8
#define VROW    520                  // bf16 tile row pad -> conflict-free b128 reads
#define VROWF   516                  // f32 tile row pad (stride%64==4 dwords)
#define LROW    40

typedef __attribute__((ext_vector_type(16))) __bf16 v16bf;
typedef __attribute__((ext_vector_type(8)))  __bf16 v8bf;
typedef __attribute__((ext_vector_type(4)))  __bf16 v4bf;
typedef __attribute__((ext_vector_type(8)))  float  v8f;
typedef __attribute__((ext_vector_type(4)))  int    v4i;

union U16bf { v16bf v; v8bf h[2]; };

// ---------------- CDNA5 async HBM->LDS copy (ASYNCcnt path) ----------------
#ifndef __has_builtin
#define __has_builtin(x) 0
#endif
#if __has_builtin(__builtin_amdgcn_global_load_async_to_lds_b128)
#define HAVE_ASYNC 1
#else
#define HAVE_ASYNC 0
#endif

__device__ __forceinline__ void async_copy_b128(void* lds, const void* g) {
#if HAVE_ASYNC
  __builtin_amdgcn_global_load_async_to_lds_b128(
      (__attribute__((address_space(1))) v4i*)g,
      (__attribute__((address_space(3))) v4i*)lds, 0, 0);
#else
  *(float4*)lds = *(const float4*)g;   // synchronous fallback
#endif
}

template <int N>
__device__ __forceinline__ void async_wait() {
#if HAVE_ASYNC
#if __has_builtin(__builtin_amdgcn_s_wait_asynccnt)
  __builtin_amdgcn_s_wait_asynccnt(N);
#else
  asm volatile("s_wait_asynccnt %0" :: "n"(N));
#endif
#endif
}

// ---------------- dynamic LDS layout (total ~174 KB, WGP allows 320 KB) ----
constexpr unsigned SZ_F    = T_TILE * VROWF * 4;       // 66048
constexpr unsigned OFF_F0  = 0;
constexpr unsigned OFF_F1  = OFF_F0 + SZ_F;            // 66048
constexpr unsigned OFF_SB  = OFF_F1 + SZ_F;            // 132096
constexpr unsigned SZ_SB   = T_TILE * VROW * 2;        // 33280
constexpr unsigned OFF_LOC = OFF_SB + SZ_SB;           // 165376
constexpr unsigned SZ_LOC  = T_TILE * LROW * 2;        // 2560
constexpr unsigned OFF_CUM = OFF_LOC + SZ_LOC;         // 167936
constexpr unsigned OFF_CW  = OFF_CUM + 64 * 4;         // 168192
constexpr unsigned OFF_Q   = OFF_CW + LOC * KW * 4;    // 172160
constexpr unsigned OFF_VW  = OFF_Q + ATT * 4;          // 172672
constexpr unsigned OFF_EP  = OFF_VW + ATT * 4;         // 173184
constexpr unsigned OFF_WT  = OFF_EP + 8 * T_TILE * 4;  // 174208
constexpr unsigned OFF_ST  = OFF_WT + T_TILE * 4;      // 174336
constexpr unsigned SMEM_TOTAL = OFF_ST + 16;           // 174352

// ---------------------------------------------------------------------------
// Prep 1: q = query @ Q_w.T + Q_b   (64x128, trivial)
// ---------------------------------------------------------------------------
__global__ __launch_bounds__(128) void prep_q_kernel(
    const float* __restrict__ query, const float* __restrict__ Qw,
    const float* __restrict__ Qb, float* __restrict__ qBuf)
{
  const int b = blockIdx.x;
  const int a = threadIdx.x;
  const float4* q4 = (const float4*)(query + (size_t)b * DEC);
  const float4* w4 = (const float4*)(Qw + (size_t)a * DEC);
  float acc = 0.f;
  #pragma unroll 8
  for (int j = 0; j < DEC / 4; ++j) {
    float4 x = q4[j], y = w4[j];
    acc += x.x * y.x + x.y * y.y + x.z * y.z + x.w * y.w;
  }
  qBuf[b * ATT + a] = acc + Qb[a];
}

// ---------------------------------------------------------------------------
// Prep 2: convert V_w (128x512) and loc_proj_w (128x32) to bf16, [a][e] layout
// ---------------------------------------------------------------------------
__global__ __launch_bounds__(256) void prep_cvt_kernel(
    const float* __restrict__ Vw, const float* __restrict__ locProj,
    __bf16* __restrict__ VwBf, __bf16* __restrict__ locBf)
{
  int i = blockIdx.x * 256 + threadIdx.x;
  if (i < ATT * ENC) VwBf[i] = (__bf16)Vw[i];
  if (i < ATT * LOC) locBf[i] = (__bf16)locProj[i];
}

// ---------------------------------------------------------------------------
// Main fused kernel: per (batch, T-chunk); flash-softmax over tiles of 32 t.
// Double-buffered async HBM->LDS tile copies overlap WMMA + softmax.
// ---------------------------------------------------------------------------
__global__ __launch_bounds__(256) void attn_main_kernel(
    const float* __restrict__ values,
    const float* __restrict__ cum,
    const unsigned char* __restrict__ maskp,
    const float* __restrict__ qBuf,
    const float* __restrict__ vwRow,
    const float* __restrict__ convW,
    const __bf16* __restrict__ VwBf,
    const __bf16* __restrict__ locBf,
    float* __restrict__ eBuf,
    float* __restrict__ mPart, float* __restrict__ sPart,
    float* __restrict__ ctxPart)
{
  extern __shared__ __align__(16) char smem[];
  float* fbuf0 = (float*)(smem + OFF_F0);
  float* fbuf1 = (float*)(smem + OFF_F1);
  __bf16 (*sB)[VROW]   = reinterpret_cast<__bf16 (*)[VROW]>(smem + OFF_SB);
  __bf16 (*sLoc)[LROW] = reinterpret_cast<__bf16 (*)[LROW]>(smem + OFF_LOC);
  float* sCum   = (float*)(smem + OFF_CUM);
  float* sConvW = (float*)(smem + OFF_CW);
  float* sQ     = (float*)(smem + OFF_Q);
  float* sVw    = (float*)(smem + OFF_VW);
  float (*sEpart)[T_TILE] = reinterpret_cast<float (*)[T_TILE]>(smem + OFF_EP);
  float* sWt    = (float*)(smem + OFF_WT);
  float* sStat  = (float*)(smem + OFF_ST);   // [0]=m, [1]=s, [2]=scale

  const int b     = blockIdx.x;
  const int chunk = blockIdx.y;
  const int tid   = threadIdx.x;
  const int wv    = tid >> 5;
  const int lane  = tid & 31;
  const int nn    = lane & 15;     // fragment row/col within 16
  const int half  = lane >> 4;     // selects K-halves per ISA 16-bit layout
  const int aCol  = wv * 16 + nn;  // this wave covers ATT columns [16w,16w+16)
  const int koff  = half * 8;
  const int tBase = chunk * T_CHUNK;

  // ---- prologue: issue async copy of tile 0, load small constants ----
  {
    const float4* g0 = (const float4*)(values + ((size_t)b * TT_ + tBase) * ENC);
    #pragma unroll
    for (int i = 0; i < 16; ++i) {
      int idx = tid + i * 256;           // 0..4095 16B chunks
      int r   = idx >> 7;
      int c4  = idx & 127;
      async_copy_b128(&fbuf0[r * VROWF + c4 * 4], &g0[(size_t)r * (ENC / 4) + c4]);
    }
  }
  for (int i = tid; i < LOC * KW; i += 256) sConvW[i] = convW[i];
  if (tid < ATT) { sQ[tid] = qBuf[b * ATT + tid]; sVw[tid] = vwRow[tid]; }
  if (tid == 0) { sStat[0] = -1e30f; sStat[1] = 0.0f; }

  float ctxA = 0.f, ctxB = 0.f;          // each thread owns e = tid, tid+256
  const int e0 = tid, e1 = tid + 256;

  for (int tile = 0; tile < N_TILES; ++tile) {
    const int t0 = tBase + tile * T_TILE;
    float* fcur = (tile & 1) ? fbuf1 : fbuf0;
    float* fnxt = (tile & 1) ? fbuf0 : fbuf1;

    // ---- issue async copy of next tile, then wait for current tile ----
    if (tile + 1 < N_TILES) {
      const float4* gn =
          (const float4*)(values + ((size_t)b * TT_ + t0 + T_TILE) * ENC);
      #pragma unroll
      for (int i = 0; i < 16; ++i) {
        int idx = tid + i * 256;
        int r   = idx >> 7;
        int c4  = idx & 127;
        async_copy_b128(&fnxt[r * VROWF + c4 * 4], &gn[(size_t)r * (ENC / 4) + c4]);
      }
      async_wait<16>();                  // in-order: older 16 (current tile) done
    } else {
      async_wait<0>();
    }
    if (tid < T_TILE + KW - 1) {         // cum-attn halo for SAME conv
      int g = t0 - PADW + tid;
      sCum[tid] = (g >= 0 && g < TT_) ? cum[(size_t)b * TT_ + g] : 0.0f;
    }
    __syncthreads();                                    // (W) tile resident

    // ---- shared f32 -> bf16 conversion pass (once, not per wave) ----
    #pragma unroll
    for (int i = 0; i < 16; ++i) {
      int idx = tid + i * 256;
      int r   = idx >> 7;
      int c4  = idx & 127;
      float4 v = *(const float4*)&fcur[r * VROWF + c4 * 4];
      v4bf p;
      p[0] = (__bf16)v.x; p[1] = (__bf16)v.y; p[2] = (__bf16)v.z; p[3] = (__bf16)v.w;
      *(v4bf*)&sB[r][c4 * 4] = p;
    }
    // ---- conv (32 filters x K=31, cross-correlation, SAME) ----
    #pragma unroll
    for (int i = 0; i < 4; ++i) {
      int idx = tid + i * 256;           // 0..1023 = (c,t)
      int c = idx >> 5;
      int t = idx & 31;
      float acc = 0.f;
      #pragma unroll
      for (int k = 0; k < KW; ++k) acc += sConvW[c * KW + k] * sCum[t + k];
      sLoc[t][c] = (__bf16)acc;
    }
    __syncthreads();                                    // (X)

    // ---- WMMA: energies GEMM, acc = loc_proj (K=32) + values@V_w.T (K=512)
    v8f acc0 = {0,0,0,0,0,0,0,0};        // t rows 0..15
    v8f acc1 = {0,0,0,0,0,0,0,0};        // t rows 16..31
    {
      U16bf afr, bfr;
      const v8bf* bRow = (const v8bf*)(locBf + (size_t)aCol * LOC);
      bfr.h[0] = bRow[koff >> 3];
      bfr.h[1] = bRow[(koff + 16) >> 3];
      const v8bf* aR0 = (const v8bf*)&sLoc[nn][0];
      const v8bf* aR1 = (const v8bf*)&sLoc[16 + nn][0];
      afr.h[0] = aR0[koff >> 3]; afr.h[1] = aR0[(koff + 16) >> 3];
      acc0 = __builtin_amdgcn_wmma_f32_16x16x32_bf16(false, afr.v, false, bfr.v,
                                                     (short)0, acc0, false, false);
      afr.h[0] = aR1[koff >> 3]; afr.h[1] = aR1[(koff + 16) >> 3];
      acc1 = __builtin_amdgcn_wmma_f32_16x16x32_bf16(false, afr.v, false, bfr.v,
                                                     (short)0, acc1, false, false);
    }
    {
      const v8bf* bRow = (const v8bf*)(VwBf + (size_t)aCol * ENC);
      const v8bf* aR0  = (const v8bf*)&sB[nn][0];
      const v8bf* aR1  = (const v8bf*)&sB[16 + nn][0];
      #pragma unroll
      for (int ks = 0; ks < ENC / 32; ++ks) {
        int kb = ks * 32;
        U16bf a0f, a1f, bff;
        bff.h[0] = bRow[(kb + koff) >> 3]; bff.h[1] = bRow[(kb + koff + 16) >> 3];
        a0f.h[0] = aR0[(kb + koff) >> 3];  a0f.h[1] = aR0[(kb + koff + 16) >> 3];
        a1f.h[0] = aR1[(kb + koff) >> 3];  a1f.h[1] = aR1[(kb + koff + 16) >> 3];
        acc0 = __builtin_amdgcn_wmma_f32_16x16x32_bf16(false, a0f.v, false, bff.v,
                                                       (short)0, acc0, false, false);
        acc1 = __builtin_amdgcn_wmma_f32_16x16x32_bf16(false, a1f.v, false, bff.v,
                                                       (short)0, acc1, false, false);
      }
    }
    // ---- epilogue: tanh(acc+q)*v_w, reduce over 16 a-cols (deterministic)
    {
      float qv = sQ[aCol];
      float vv = sVw[aCol];
      #pragma unroll
      for (int r = 0; r < 8; ++r) {
        float ea = tanhf(acc0[r] + qv) * vv;
        ea += __shfl_xor(ea, 1); ea += __shfl_xor(ea, 2);
        ea += __shfl_xor(ea, 4); ea += __shfl_xor(ea, 8);
        float eb = tanhf(acc1[r] + qv) * vv;
        eb += __shfl_xor(eb, 1); eb += __shfl_xor(eb, 2);
        eb += __shfl_xor(eb, 4); eb += __shfl_xor(eb, 8);
        if (nn == 0) {                   // lanes 0 and 16 hold half-sums
          sEpart[wv][half * 8 + r]      = ea;   // t = half*8+r  (rows 0..15)
          sEpart[wv][16 + half * 8 + r] = eb;   // t = 16+...    (rows 16..31)
        }
      }
    }
    __syncthreads();                                    // (Y)

    // ---- flash softmax update (wave 0 only) ----
    if (tid < T_TILE) {
      float e = sEpart[0][tid];
      #pragma unroll
      for (int j = 1; j < 8; ++j) e += sEpart[j][tid];
      if (maskp[(size_t)b * TT_ + t0 + tid]) e = -1e30f;
      eBuf[(size_t)b * TT_ + t0 + tid] = e;      // raw energies for pass 2
      float m = e;
      m = fmaxf(m, __shfl_xor(m, 16)); m = fmaxf(m, __shfl_xor(m, 8));
      m = fmaxf(m, __shfl_xor(m, 4));  m = fmaxf(m, __shfl_xor(m, 2));
      m = fmaxf(m, __shfl_xor(m, 1));
      float oldM = sStat[0];
      float newM = fmaxf(oldM, m);
      float p = __expf(e - newM);
      sWt[tid] = p;
      float ps = p;
      ps += __shfl_xor(ps, 16); ps += __shfl_xor(ps, 8);
      ps += __shfl_xor(ps, 4);  ps += __shfl_xor(ps, 2); ps += __shfl_xor(ps, 1);
      if (tid == 0) {
        float sc = __expf(oldM - newM);
        sStat[2] = sc;
        sStat[0] = newM;
        sStat[1] = sStat[1] * sc + ps;
      }
    }
    __syncthreads();                                    // (Z)

    // ---- context accumulation from the f32 tile ----
    {
      float sc = sStat[2];
      ctxA *= sc; ctxB *= sc;
      #pragma unroll
      for (int t = 0; t < T_TILE; ++t) {
        float wt = sWt[t];
        ctxA += wt * fcur[t * VROWF + e0];
        ctxB += wt * fcur[t * VROWF + e1];
      }
    }
    __syncthreads();                                    // (E) protects buffers
  }

  const int pIdx = b * S_SPLIT + chunk;
  if (tid == 0) { mPart[pIdx] = sStat[0]; sPart[pIdx] = sStat[1]; }
  ctxPart[(size_t)pIdx * ENC + e0] = ctxA;
  ctxPart[(size_t)pIdx * ENC + e1] = ctxB;
}

// ---------------------------------------------------------------------------
// Combine: merge S_SPLIT partial softmax states per batch; emit outputs.
// d_out = [context (B x ENC) | weights (B x T)]
// ---------------------------------------------------------------------------
__global__ __launch_bounds__(256) void attn_combine_kernel(
    const float* __restrict__ mPart, const float* __restrict__ sPart,
    const float* __restrict__ ctxPart, const float* __restrict__ eBuf,
    float* __restrict__ outCtx, float* __restrict__ outW)
{
  __shared__ float sSc[S_SPLIT];
  __shared__ float sM, sInv;
  const int b = blockIdx.x;
  const int tid = threadIdx.x;
  if (tid == 0) {
    float M = -1e30f;
    for (int j = 0; j < S_SPLIT; ++j) M = fmaxf(M, mPart[b * S_SPLIT + j]);
    float s = 0.f;
    for (int j = 0; j < S_SPLIT; ++j) {
      float sc = __expf(mPart[b * S_SPLIT + j] - M);
      sSc[j] = sc;
      s += sPart[b * S_SPLIT + j] * sc;
    }
    sM = M;
    sInv = 1.0f / s;
  }
  __syncthreads();
  float M = sM, inv = sInv;
  for (int e = tid; e < ENC; e += 256) {
    float acc = 0.f;
    #pragma unroll
    for (int j = 0; j < S_SPLIT; ++j)
      acc += ctxPart[(size_t)(b * S_SPLIT + j) * ENC + e] * sSc[j];
    outCtx[(size_t)b * ENC + e] = acc * inv;
  }
  for (int t = tid; t < TT_; t += 256)
    outW[(size_t)b * TT_ + t] = __expf(eBuf[(size_t)b * TT_ + t] - M) * inv;
}

// ---------------------------------------------------------------------------
extern "C" void kernel_launch(void* const* d_in, const int* in_sizes, int n_in,
                              void* d_out, int out_size, void* d_ws, size_t ws_size,
                              hipStream_t stream)
{
  const float*         query   = (const float*)d_in[0];
  const float*         values  = (const float*)d_in[1];
  const float*         cum     = (const float*)d_in[2];
  const unsigned char* maskp   = (const unsigned char*)d_in[3];
  const float*         Qw      = (const float*)d_in[4];
  const float*         Qb      = (const float*)d_in[5];
  const float*         Vw      = (const float*)d_in[6];
  const float*         convW   = (const float*)d_in[7];
  const float*         locProj = (const float*)d_in[8];
  const float*         vwRow   = (const float*)d_in[9];
  (void)in_sizes; (void)n_in; (void)out_size; (void)ws_size;

  char* ws = (char*)d_ws;
  size_t off = 0;
  float*  qBuf    = (float*)(ws + off); off += (size_t)BB * ATT * 4;            // 32 KB
  float*  eBuf    = (float*)(ws + off); off += (size_t)BB * TT_ * 4;            // 512 KB
  float*  mPart   = (float*)(ws + off); off += (size_t)BB * S_SPLIT * 4;
  float*  sPart   = (float*)(ws + off); off += (size_t)BB * S_SPLIT * 4;
  float*  ctxPart = (float*)(ws + off); off += (size_t)BB * S_SPLIT * ENC * 4;  // 1 MB
  __bf16* VwBf    = (__bf16*)(ws + off); off += (size_t)ATT * ENC * 2;          // 128 KB
  __bf16* locBf   = (__bf16*)(ws + off); off += (size_t)ATT * LOC * 2;

  prep_q_kernel<<<dim3(BB), dim3(ATT), 0, stream>>>(query, Qw, Qb, qBuf);
  prep_cvt_kernel<<<dim3((ATT * ENC + 255) / 256), dim3(256), 0, stream>>>(
      Vw, locProj, VwBf, locBf);
  attn_main_kernel<<<dim3(BB, S_SPLIT), dim3(256), SMEM_TOTAL, stream>>>(
      values, cum, maskp, qBuf, vwRow, convW, VwBf, locBf,
      eBuf, mPart, sPart, ctxPart);
  attn_combine_kernel<<<dim3(BB), dim3(256), 0, stream>>>(
      mPart, sPart, ctxPart, eBuf, (float*)d_out, (float*)d_out + (size_t)BB * ENC);
}
